// MultiHeadAttention_64733747085699
// MI455X (gfx1250) — matrix operand order repaired and checked by reading, not compiled, not run on hardware
//
#include <hip/hip_runtime.h>

// Problem constants (fixed by the reference)
#define BATCH   4
#define SEQ     2048
#define DMODEL  1024
#define NHEADS  16
#define DHEAD   64
#define MTOK    (BATCH * SEQ)   // 8192 tokens

typedef __bf16 bf16_t;
typedef __attribute__((ext_vector_type(16))) __bf16 bf16x16;
typedef __attribute__((ext_vector_type(8)))  __bf16 bf16x8;
typedef __attribute__((ext_vector_type(8)))  float  floatx8;

// ---------------------------------------------------------------------------
// WMMA helper: D = A(16x32 bf16) x B(32x16 bf16) + C(16x16 f32)
// ---------------------------------------------------------------------------
static __device__ __forceinline__ floatx8 wmma_bf16(bf16x16 a, bf16x16 b, floatx8 c) {
  return __builtin_amdgcn_wmma_f32_16x16x32_bf16(false, a, false, b, (short)0, c,
                                                 false, false);
}

// A-operand (16x32, MxK) per ISA 7.12.2:
//   lane L (0..15): row M=L, K = 0..7 and 16..23 ; lane L+16: K = 8..15 and 24..31
static __device__ __forceinline__ bf16x16 load_a_frag(const bf16_t* base,
                                                      int rowStride, int lane) {
  const bf16_t* p = base + (size_t)(lane & 15) * rowStride + ((lane >> 4) << 3);
  bf16x8 lo = *reinterpret_cast<const bf16x8*>(p);
  bf16x8 hi = *reinterpret_cast<const bf16x8*>(p + 16);
  return __builtin_shufflevector(lo, hi, 0,1,2,3,4,5,6,7,8,9,10,11,12,13,14,15);
}

// B-operand (32x16, KxN), element (k,n) = base[n*rowStride + k]:
//   lane L (0..15): col N=L, K = 0..7 and 16..23 ; lane L+16: K = 8..15 and 24..31
static __device__ __forceinline__ bf16x16 load_b_frag(const bf16_t* base,
                                                      int rowStride, int lane) {
  const bf16_t* p = base + (size_t)(lane & 15) * rowStride + ((lane >> 4) << 3);
  bf16x8 lo = *reinterpret_cast<const bf16x8*>(p);
  bf16x8 hi = *reinterpret_cast<const bf16x8*>(p + 16);
  return __builtin_shufflevector(lo, hi, 0,1,2,3,4,5,6,7,8,9,10,11,12,13,14,15);
}

// gfx1250 async global->LDS copy, 16B per lane, tracked by ASYNCcnt.
// Low 32 bits of a generic shared pointer are the LDS byte address (ISA 10.2).
static __device__ __forceinline__ void async_load_b128(unsigned lds_addr,
                                                       const void* gaddr) {
  asm volatile("global_load_async_to_lds_b128 %0, %1, off"
               :: "v"(lds_addr), "v"(gaddr) : "memory");
}
static __device__ __forceinline__ void wait_async_le6() {
  asm volatile("s_wait_asynccnt 0x6" ::: "memory");
}
static __device__ __forceinline__ void wait_async_0() {
  asm volatile("s_wait_asynccnt 0x0" ::: "memory");
}

// 16-lane (half-wave) reductions; xor masks < 16 keep lanes 0-15 / 16-31 separate
static __device__ __forceinline__ float hred_max(float v) {
  v = fmaxf(v, __shfl_xor(v, 1, 32));
  v = fmaxf(v, __shfl_xor(v, 2, 32));
  v = fmaxf(v, __shfl_xor(v, 4, 32));
  v = fmaxf(v, __shfl_xor(v, 8, 32));
  return v;
}
static __device__ __forceinline__ float hred_sum(float v) {
  v += __shfl_xor(v, 1, 32);
  v += __shfl_xor(v, 2, 32);
  v += __shfl_xor(v, 4, 32);
  v += __shfl_xor(v, 8, 32);
  return v;
}

// ---------------------------------------------------------------------------
// fp32 -> bf16 convert (grid-stride)
// ---------------------------------------------------------------------------
__global__ void cvt_f32_bf16(const float* __restrict__ in, bf16_t* __restrict__ out,
                             unsigned n) {
  unsigned i = blockIdx.x * blockDim.x + threadIdx.x;
  unsigned stride = gridDim.x * blockDim.x;
  for (; i < n; i += stride) out[i] = (bf16_t)in[i];
}

// ---------------------------------------------------------------------------
// bf16 WMMA GEMM: C[M,N] = A[M,K] * W[N,K]^T + bias
//   4 waves / workgroup, workgroup tile 128(M) x 64(N); wave tile 32x64.
//   A (128x32) and B (64x32) k-slabs ping-pong buffered in LDS, filled with
//   global_load_async_to_lds_b128 (ASYNCcnt) and consumed via ds loads.
//   K-loop manually unrolled 2x so buffer indices are static.
//   mode 0: store bf16 as [b, h, s, d]   (Q, K)
//   mode 1: store bf16 as [b, h, d, s]   (V transposed)
//   mode 2: store fp32 row-major [M, N]  (final output projection)
// ---------------------------------------------------------------------------
struct GemmSmem {
  bf16_t A[2][128 * 32];  // 8 KB per buffer
  bf16_t B[2][64 * 32];   // 4 KB per buffer
};

static __device__ __forceinline__ void stage_slab(GemmSmem* sm, int buf,
                                                  const bf16_t* A, const bf16_t* W,
                                                  int m0, int n0, int K, int k0,
                                                  int tid) {
  // A slab: 128 rows x 32 cols bf16 = 512 x 16B chunks; 4 per thread
#pragma unroll
  for (int i = 0; i < 4; ++i) {
    int c = tid + i * 128;
    int row = c >> 2, seg = (c & 3) << 3;            // seg in elements
    const bf16_t* g = A + (size_t)(m0 + row) * K + k0 + seg;
    unsigned l = (unsigned)(size_t)&sm->A[buf][row * 32 + seg];
    async_load_b128(l, g);
  }
  // B slab: 64 rows x 32 cols = 256 chunks; 2 per thread
#pragma unroll
  for (int i = 0; i < 2; ++i) {
    int c = tid + i * 128;
    int row = c >> 2, seg = (c & 3) << 3;
    const bf16_t* g = W + (size_t)(n0 + row) * K + k0 + seg;
    unsigned l = (unsigned)(size_t)&sm->B[buf][row * 32 + seg];
    async_load_b128(l, g);
  }
}

__global__ __launch_bounds__(128)
void gemm_bf16_wmma(const bf16_t* __restrict__ A, const bf16_t* __restrict__ W,
                    const float* __restrict__ bias, void* __restrict__ out,
                    int M, int N, int K, int mode) {
  __shared__ GemmSmem sm;
  const int tid = threadIdx.x;
  const int lane = tid & 31;
  const int wave = tid >> 5;
  const int m0 = blockIdx.x * 128;
  const int n0 = blockIdx.y * 64;

  floatx8 acc[2][4];
#pragma unroll
  for (int i = 0; i < 2; ++i)
#pragma unroll
    for (int j = 0; j < 4; ++j) acc[i][j] = (floatx8)(0.0f);

  // all fragment loads first (one dscnt wait), then the 8-WMMA chain
  auto compute = [&](int buf) {
    const bf16_t* As = sm.A[buf] + (wave * 32) * 32;
    const bf16_t* Bs = sm.B[buf];
    bf16x16 a0 = load_a_frag(As, 32, lane);
    bf16x16 a1 = load_a_frag(As + 16 * 32, 32, lane);
    bf16x16 b0 = load_b_frag(Bs + 0 * 16 * 32, 32, lane);
    bf16x16 b1 = load_b_frag(Bs + 1 * 16 * 32, 32, lane);
    bf16x16 b2 = load_b_frag(Bs + 2 * 16 * 32, 32, lane);
    bf16x16 b3 = load_b_frag(Bs + 3 * 16 * 32, 32, lane);
    acc[0][0] = wmma_bf16(a0, b0, acc[0][0]);
    acc[1][0] = wmma_bf16(a1, b0, acc[1][0]);
    acc[0][1] = wmma_bf16(a0, b1, acc[0][1]);
    acc[1][1] = wmma_bf16(a1, b1, acc[1][1]);
    acc[0][2] = wmma_bf16(a0, b2, acc[0][2]);
    acc[1][2] = wmma_bf16(a1, b2, acc[1][2]);
    acc[0][3] = wmma_bf16(a0, b3, acc[0][3]);
    acc[1][3] = wmma_bf16(a1, b3, acc[1][3]);
  };

  stage_slab(&sm, 0, A, W, m0, n0, K, 0, tid);

  for (int k0 = 0; k0 < K; k0 += 64) {
    // stage k0+32 into buf1 (safe: buf1 readers finished in previous round)
    stage_slab(&sm, 1, A, W, m0, n0, K, k0 + 32, tid);
    wait_async_le6();          // buf0's 6 loads done; buf1's 6 in flight
    __syncthreads();
    compute(0);
    __syncthreads();

    if (k0 + 64 < K) {
      stage_slab(&sm, 0, A, W, m0, n0, K, k0 + 64, tid);
      wait_async_le6();
    } else {
      wait_async_0();
    }
    __syncthreads();
    compute(1);
    __syncthreads();
  }

  // Epilogue. C layout: lane -> n_local = lane&15, half = lane>>4; vgpr r -> row
  const int nl = lane & 15, half = lane >> 4;
  const int mw = m0 + wave * 32;
#pragma unroll
  for (int mi = 0; mi < 2; ++mi) {
#pragma unroll
    for (int nj = 0; nj < 4; ++nj) {
#pragma unroll
      for (int r = 0; r < 8; ++r) {
        int m = mw + mi * 16 + r + 8 * half;
        int n = n0 + nj * 16 + nl;
        float v = acc[mi][nj][r] + bias[n];
        if (mode == 2) {
          ((float*)out)[(size_t)m * N + n] = v;
        } else {
          int b = m >> 11, s = m & (SEQ - 1);
          int h = n >> 6,  d = n & (DHEAD - 1);
          bf16_t bv = (bf16_t)v;
          if (mode == 0)
            ((bf16_t*)out)[(((size_t)b * NHEADS + h) * SEQ + s) * DHEAD + d] = bv;
          else
            ((bf16_t*)out)[(((size_t)b * NHEADS + h) * DHEAD + d) * SEQ + s] = bv;
        }
      }
    }
  }
}

// ---------------------------------------------------------------------------
// Causal flash attention, one wave per (b*h, 16-query tile).
//   Q,K: [b,h,s,64] bf16 ; Vt: [b,h,64,s] bf16 ; O: [b,s,1024] bf16
// ---------------------------------------------------------------------------
__global__ __launch_bounds__(32)
void attn_wmma(const bf16_t* __restrict__ Q, const bf16_t* __restrict__ Km,
               const bf16_t* __restrict__ Vt, bf16_t* __restrict__ O) {
  __shared__ __align__(32) bf16_t pbuf[16 * 32];   // P tile staging (C->A relayout)

  const int lane = threadIdx.x;
  const int q0 = blockIdx.x * 16;
  const int bh = blockIdx.y;                        // b*NHEADS + h
  const int nl = lane & 15, half = lane >> 4;

  const bf16_t* Qb = Q  + (size_t)bh * SEQ * DHEAD;
  const bf16_t* Kb = Km + (size_t)bh * SEQ * DHEAD;
  const bf16_t* Vb = Vt + (size_t)bh * DHEAD * SEQ;

  bf16x16 aq0 = load_a_frag(Qb + (size_t)q0 * DHEAD + 0,  DHEAD, lane);
  bf16x16 aq1 = load_a_frag(Qb + (size_t)q0 * DHEAD + 32, DHEAD, lane);

  floatx8 acc[4];
#pragma unroll
  for (int t = 0; t < 4; ++t) acc[t] = (floatx8)(0.0f);
  float rmax[8], rsum[8];
#pragma unroll
  for (int r = 0; r < 8; ++r) { rmax[r] = -3.0e38f; rsum[r] = 0.0f; }

  const float scale = 0.125f;                       // 1/sqrt(64)
  const int jmax = q0 + 16;                         // causal bound

  for (int j0 = 0; j0 < jmax; j0 += 32) {
    if (j0 + 32 < jmax) {                           // prefetch next K/V tiles
      __builtin_prefetch(Kb + (size_t)(j0 + 32) * DHEAD, 0, 3);
      __builtin_prefetch(Vb + (size_t)(16 * (lane & 3)) * SEQ + j0 + 32, 0, 3);
    }
    // ---- scores: two 16x16 tiles over keys [j0,j0+16) and [j0+16,j0+32)
    floatx8 s0 = (floatx8)(0.0f), s1 = (floatx8)(0.0f);
    {
      bf16x16 bk00 = load_b_frag(Kb + (size_t)j0 * DHEAD + 0,         DHEAD, lane);
      bf16x16 bk01 = load_b_frag(Kb + (size_t)j0 * DHEAD + 32,        DHEAD, lane);
      bf16x16 bk10 = load_b_frag(Kb + (size_t)(j0 + 16) * DHEAD + 0,  DHEAD, lane);
      bf16x16 bk11 = load_b_frag(Kb + (size_t)(j0 + 16) * DHEAD + 32, DHEAD, lane);
      s0 = wmma_bf16(aq0, bk00, s0);
      s0 = wmma_bf16(aq1, bk01, s0);
      s1 = wmma_bf16(aq0, bk10, s1);
      s1 = wmma_bf16(aq1, bk11, s1);
    }

    const bool needmask = (j0 + 31 > q0);           // only diagonal blocks
    float corr[8];
#pragma unroll
    for (int r = 0; r < 8; ++r) {
      int qrow = q0 + r + 8 * half;
      float x0 = s0[r] * scale;
      float x1 = s1[r] * scale;
      if (needmask) {
        if (j0 + nl      > qrow) x0 = -3.0e38f;
        if (j0 + 16 + nl > qrow) x1 = -3.0e38f;
      }
      float mx = hred_max(fmaxf(x0, x1));
      float nm = fmaxf(rmax[r], mx);
      corr[r]  = __expf(rmax[r] - nm);
      rmax[r]  = nm;
      float p0 = __expf(x0 - nm);
      float p1 = __expf(x1 - nm);
      rsum[r]  = rsum[r] * corr[r] + hred_sum(p0 + p1);
      s0[r] = p0;
      s1[r] = p1;
    }

#pragma unroll
    for (int t = 0; t < 4; ++t)
#pragma unroll
      for (int r = 0; r < 8; ++r) acc[t][r] *= corr[r];

    // ---- P (16x32) through LDS: C-layout -> A-layout
#pragma unroll
    for (int r = 0; r < 8; ++r) {
      int m = r + 8 * half;
      pbuf[m * 32 + nl]      = (bf16_t)s0[r];
      pbuf[m * 32 + 16 + nl] = (bf16_t)s1[r];
    }
    asm volatile("s_wait_dscnt 0" ::: "memory");
    bf16x16 pf = load_a_frag(pbuf, 32, lane);

    // ---- O += P x V  (Vt rows are head-dim, contiguous over keys)
#pragma unroll
    for (int t = 0; t < 4; ++t) {
      bf16x16 bv = load_b_frag(Vb + (size_t)(t * 16) * SEQ + j0, SEQ, lane);
      acc[t] = wmma_bf16(pf, bv, acc[t]);
    }
  }

  // ---- epilogue: normalize and store to [b, s, h*64 + d] bf16
  const int b = bh >> 4, h = bh & (NHEADS - 1);
#pragma unroll
  for (int r = 0; r < 8; ++r) {
    float inv = 1.0f / rsum[r];
    int tok = q0 + r + 8 * half;
#pragma unroll
    for (int t = 0; t < 4; ++t) {
      float v = acc[t][r] * inv;
      O[((size_t)b * SEQ + tok) * DMODEL + h * DHEAD + t * 16 + nl] = (bf16_t)v;
    }
  }
}

// ---------------------------------------------------------------------------
// Host-side orchestration
// ---------------------------------------------------------------------------
extern "C" void kernel_launch(void* const* d_in, const int* in_sizes, int n_in,
                              void* d_out, int out_size, void* d_ws, size_t ws_size,
                              hipStream_t stream) {
  const float* x  = (const float*)d_in[0];
  // d_in[1] = causal mask (bool) -- computed analytically instead
  const float* Wq = (const float*)d_in[2];
  const float* bq = (const float*)d_in[3];
  const float* Wk = (const float*)d_in[4];
  const float* bk = (const float*)d_in[5];
  const float* Wv = (const float*)d_in[6];
  const float* bv = (const float*)d_in[7];
  const float* Wo = (const float*)d_in[8];
  const float* bo = (const float*)d_in[9];

  char* ws = (char*)d_ws;
  size_t off = 0;
  auto alloc = [&](size_t bytes) -> void* {
    void* p = ws + off;
    off += (bytes + 255) & ~(size_t)255;
    return p;
  };

  const size_t nx = (size_t)MTOK * DMODEL;
  const size_t nw = (size_t)DMODEL * DMODEL;
  const size_t nqkv = (size_t)BATCH * NHEADS * SEQ * DHEAD;

  bf16_t* xb  = (bf16_t*)alloc(nx * 2);           // reused as attention output O
  bf16_t* wqb = (bf16_t*)alloc(nw * 2);
  bf16_t* wkb = (bf16_t*)alloc(nw * 2);
  bf16_t* wvb = (bf16_t*)alloc(nw * 2);
  bf16_t* wob = (bf16_t*)alloc(nw * 2);
  bf16_t* Qb  = (bf16_t*)alloc(nqkv * 2);
  bf16_t* Kb  = (bf16_t*)alloc(nqkv * 2);
  bf16_t* Vtb = (bf16_t*)alloc(nqkv * 2);

  cvt_f32_bf16<<<2048, 256, 0, stream>>>(x,  xb,  (unsigned)nx);
  cvt_f32_bf16<<<512,  256, 0, stream>>>(Wq, wqb, (unsigned)nw);
  cvt_f32_bf16<<<512,  256, 0, stream>>>(Wk, wkb, (unsigned)nw);
  cvt_f32_bf16<<<512,  256, 0, stream>>>(Wv, wvb, (unsigned)nw);
  cvt_f32_bf16<<<512,  256, 0, stream>>>(Wo, wob, (unsigned)nw);

  dim3 ggrid(MTOK / 128, DMODEL / 64);
  gemm_bf16_wmma<<<ggrid, 128, 0, stream>>>(xb, wqb, bq, Qb,  MTOK, DMODEL, DMODEL, 0);
  gemm_bf16_wmma<<<ggrid, 128, 0, stream>>>(xb, wkb, bk, Kb,  MTOK, DMODEL, DMODEL, 0);
  gemm_bf16_wmma<<<ggrid, 128, 0, stream>>>(xb, wvb, bv, Vtb, MTOK, DMODEL, DMODEL, 1);

  bf16_t* Ob = xb;
  dim3 agrid(SEQ / 16, BATCH * NHEADS);
  attn_wmma<<<agrid, 32, 0, stream>>>(Qb, Kb, Vtb, Ob);

  gemm_bf16_wmma<<<ggrid, 128, 0, stream>>>(Ob, wob, bo, d_out, MTOK, DMODEL, DMODEL, 2);
}
